// WeightformerSelfAttention_4166118277664
// MI455X (gfx1250) — matrix operand order repaired and checked by reading, not compiled, not run on hardware
//
#include <hip/hip_runtime.h>
#include <hip/hip_bf16.h>

// ---------------------------------------------------------------------------
// Banded (Longformer-style) self-attention for MI455X (gfx1250, wave32, WMMA)
// B=2, S=4096, E=1024, H=16, D=64, W=256 (one-sided).
// All matmuls: v_wmma_f32_16x16x32_bf16. Staging: global_load_async_to_lds_b128,
// double-buffered in the projection GEMM (copy overlaps WMMA).
// ---------------------------------------------------------------------------

typedef __attribute__((ext_vector_type(16))) __bf16          v16bf;
typedef __attribute__((ext_vector_type(16))) unsigned short  v16u16;
typedef __attribute__((ext_vector_type(8)))  unsigned short  v8u16;
typedef __attribute__((ext_vector_type(8)))  float           v8f;

#define BB   2
#define SS   4096
#define EE   1024
#define HH   16
#define DD   64
#define WW   256
#define NCC  (SS / WW)   // 16 query chunks
#define KWIN (3 * WW)    // 768 keys per window

// LDS row strides (elements): 16B-aligned and bank-conflict-free for frag reads
#define KROW 72    // Kt rows (64 used)
#define QROW 72    // Qt rows (64 used)
#define VROW 776   // VtT rows (768 used)
#define PROW 40    // Ps rows  (32 used)
#define GROW 40    // GEMM tile rows (32 used)

#define GEMM_M 128        // block tile M
#define GEMM_N 64         // block tile N
#define NKT    (EE / 32)  // 32 K-tiles

__device__ __forceinline__ unsigned short f2bf(float f) {
  unsigned int u = __float_as_uint(f);
  unsigned int r = u + 0x7FFFu + ((u >> 16) & 1u);   // round-to-nearest-even
  return (unsigned short)(r >> 16);
}

// LDS byte address of a __shared__ object: generic LDS pointers carry the LDS
// offset in their low 32 bits (ISA 10.2 aperture mapping).
__device__ __forceinline__ unsigned ldsOff(const void* p) {
  return (unsigned)(unsigned long long)p;
}

// CDNA5 async copy: LDS[lds_off] <- 16 bytes at gptr (per active lane).
__device__ __forceinline__ void async_b128(unsigned lds_off, const void* gptr) {
  unsigned long long ga = (unsigned long long)gptr;
  asm volatile("global_load_async_to_lds_b128 %0, %1, off"
               :: "v"(lds_off), "v"(ga) : "memory");
}
__device__ __forceinline__ void wait_async0() {
  asm volatile("s_wait_asynccnt 0x0" ::: "memory");
}
__device__ __forceinline__ void wait_async3() {
  asm volatile("s_wait_asynccnt 0x3" ::: "memory");
}

// Build a v16bf fragment from two contiguous 16-byte LDS runs.
__device__ __forceinline__ v16bf ldfrag(const unsigned short* p0,
                                        const unsigned short* p1) {
  v8u16 lo = *(const v8u16*)p0;
  v8u16 hi = *(const v8u16*)p1;
  v16u16 u = __builtin_shufflevector(lo, hi, 0, 1, 2, 3, 4, 5, 6, 7,
                                     8, 9, 10, 11, 12, 13, 14, 15);
  return __builtin_bit_cast(v16bf, u);
}

// ---------------------------------------------------------------------------
// Kernel 1: fp32 -> bf16 convert, 4 elements/thread (hidden states + weights;
// weights need no transpose: w[n][k] row-major is exactly the K-major B tile).
// ---------------------------------------------------------------------------
__global__ void wf_cvt_bf16(const float4* __restrict__ src,
                            uint2* __restrict__ dst, int n4) {
  int i = blockIdx.x * blockDim.x + threadIdx.x;
  if (i < n4) {
    float4 f = src[i];
    unsigned a = f2bf(f.x), b = f2bf(f.y), c = f2bf(f.z), d = f2bf(f.w);
    dst[i] = make_uint2(a | (b << 16), c | (d << 16));
  }
}

// ---------------------------------------------------------------------------
// Kernel 2: QKV projection GEMM, bf16 WMMA, f32 accumulate.
// grid = (M/128, N/64, 3), block = 256 threads (8 waves).
// Block tile 128x64; wave tile 32x32 (4 accumulators, 4 WMMAs per K-step).
// Double-buffered async-to-LDS staging: tile i+2's copies overlap tile i+1's
// WMMAs. V output (which==2) stored transposed to global [B,H,D,S].
// ---------------------------------------------------------------------------
__global__ void wf_qkv_gemm(const unsigned short* __restrict__ X,   // [8192,1024]
                            const unsigned short* __restrict__ Wq,  // [n][k] bf16
                            const unsigned short* __restrict__ Wk,
                            const unsigned short* __restrict__ Wv,
                            const float* __restrict__ bq,
                            const float* __restrict__ bk,
                            const float* __restrict__ bv,
                            unsigned short* __restrict__ Oq,   // [B*S, E]
                            unsigned short* __restrict__ Ok,   // [B*S, E]
                            unsigned short* __restrict__ OvT) { // [B,H,D,S]
  __shared__ __align__(16) unsigned short As[2][GEMM_M * GROW];
  __shared__ __align__(16) unsigned short Bs[2][GEMM_N * GROW];

  const int which = blockIdx.z;
  const unsigned short* W = (which == 0) ? Wq : (which == 1) ? Wk : Wv;
  const float*          bi = (which == 0) ? bq : (which == 1) ? bk : bv;
  const float scale = (which == 0) ? 0.125f : 1.0f;   // 1/sqrt(64)

  const int m0   = blockIdx.x * GEMM_M;
  const int n0   = blockIdx.y * GEMM_N;
  const int tid  = threadIdx.x;
  const int lane = tid & 31;
  const int wave = tid >> 5;            // 0..7
  const int mw   = (wave & 3) * 32;     // wave M sub-tile
  const int nw   = (wave >> 2) * 32;    // wave N sub-tile

  const int half = lane >> 4;
  const int ar   = lane & 15;
  const int akb  = half * 16;           // A: K sub-base
  const int bn   = lane & 15;
  const int bkb  = half * 16;           // B: K sub-base

  // staging chunks: A = 512 x 16B (2/thread), B = 256 x 16B (1/thread)
  const int sr0 = tid >> 2, sc0 = (tid & 3) * 8;
  const int sr1 = (tid + 256) >> 2, sc1 = ((tid + 256) & 3) * 8;

  auto stage = [&](int buf, int k0) {
    async_b128(ldsOff(&As[buf][sr0 * GROW + sc0]),
               &X[(size_t)(m0 + sr0) * EE + k0 + sc0]);
    async_b128(ldsOff(&As[buf][sr1 * GROW + sc1]),
               &X[(size_t)(m0 + sr1) * EE + k0 + sc1]);
    async_b128(ldsOff(&Bs[buf][sr0 * GROW + sc0]),
               &W[(size_t)(n0 + sr0) * EE + k0 + sc0]);
  };

  v8f acc00 = {}, acc01 = {}, acc10 = {}, acc11 = {};

  stage(0, 0);
  stage(1, 32);
  for (int it = 0; it < NKT; ++it) {
    if (it + 1 < NKT) wait_async3();    // oldest tile's 3 asyncs retired
    else              wait_async0();
    __syncthreads();

    const unsigned short* Ac = As[it & 1];
    const unsigned short* Bc = Bs[it & 1];
    const unsigned short* ar0 = Ac + (mw + ar) * GROW;
    const unsigned short* ar1 = Ac + (mw + 16 + ar) * GROW;
    const unsigned short* br0 = Bc + (nw + bn) * GROW;
    const unsigned short* br1 = Bc + (nw + 16 + bn) * GROW;
    v16bf a0 = ldfrag(ar0 + akb, ar0 + akb + 8);
    v16bf a1 = ldfrag(ar1 + akb, ar1 + akb + 8);
    v16bf b0 = ldfrag(br0 + bkb, br0 + bkb + 8);
    v16bf b1 = ldfrag(br1 + bkb, br1 + bkb + 8);

    acc00 = __builtin_amdgcn_wmma_f32_16x16x32_bf16(false, a0, false, b0,
                                                    (short)0, acc00, false, false);
    acc01 = __builtin_amdgcn_wmma_f32_16x16x32_bf16(false, a0, false, b1,
                                                    (short)0, acc01, false, false);
    acc10 = __builtin_amdgcn_wmma_f32_16x16x32_bf16(false, a1, false, b0,
                                                    (short)0, acc10, false, false);
    acc11 = __builtin_amdgcn_wmma_f32_16x16x32_bf16(false, a1, false, b1,
                                                    (short)0, acc11, false, false);
    __syncthreads();                    // all waves done reading buf (it&1)
    if (it + 2 < NKT) stage(it & 1, (it + 2) * 32);
  }

  // Epilogue. C layout: VGPR r, lane L -> M = r + 8*(L>=16), N = L%16.
  const int nn = lane & 15;
  auto storeTile = [&](const v8f& a, int mb, int nb) {
#pragma unroll
    for (int r = 0; r < 8; ++r) {
      int m = m0 + mb + r + 8 * half;           // global token row (0..8191)
      int n = n0 + nb + nn;
      unsigned short o = f2bf((a[r] + bi[n]) * scale);
      if (which == 2) {
        int b = m >> 12, s = m & (SS - 1);
        OvT[(((size_t)b * HH + (n >> 6)) * DD + (n & 63)) * SS + s] = o;
      } else {
        unsigned short* O = (which == 0) ? Oq : Ok;
        O[(size_t)m * EE + n] = o;
      }
    }
  };
  storeTile(acc00, mw, nw);
  storeTile(acc01, mw, nw + 16);
  storeTile(acc10, mw + 16, nw);
  storeTile(acc11, mw + 16, nw + 16);
}

// ---------------------------------------------------------------------------
// Kernel 3: banded flash attention.
// grid = (NC, H, B), block = 512 threads = 16 waves; wave w owns query rows
// [16w,16w+16). K window, V^T window and Q chunk async-staged into LDS.
// ---------------------------------------------------------------------------
__global__ void wf_banded_attn(const unsigned short* __restrict__ Q,  // [B,S,E]
                               const unsigned short* __restrict__ K,  // [B,S,E]
                               const unsigned short* __restrict__ VT, // [B,H,D,S]
                               const int* __restrict__ am,            // [B,S]
                               float* __restrict__ out) {             // [B,S,E]
  extern __shared__ __align__(16) unsigned short smem[];
  unsigned short* Kt  = smem;                       // [KWIN][KROW]
  unsigned short* VtT = Kt + KWIN * KROW;           // [DD][VROW]
  unsigned short* Qt  = VtT + DD * VROW;            // [WW][QROW]
  unsigned short* Ps  = Qt + WW * QROW;             // [16][16][PROW]

  const int c = blockIdx.x, h = blockIdx.y, b = blockIdx.z;
  const int tid = threadIdx.x, lane = tid & 31, wave = tid >> 5;
  const size_t baseBS = (size_t)b * SS;
  const int ecol = h * DD;

  // ---- stage K window: padded y in [0,3W), key j = c*W + y - W ----
  for (int t = tid; t < KWIN * 8; t += 512) {
    int y = t >> 3, c8 = (t & 7) * 8;
    int j = c * WW + y - WW;
    unsigned short* dst = &Kt[y * KROW + c8];
    if (j >= 0 && j < SS) {
      async_b128(ldsOff(dst), &K[(baseBS + j) * EE + ecol + c8]);
    } else {
      *(uint2*)dst       = make_uint2(0, 0);
      *(uint2*)(dst + 4) = make_uint2(0, 0);
    }
  }
  // ---- stage Q chunk (always in range) ----
  for (int t = tid; t < WW * 8; t += 512) {
    int x = t >> 3, c8 = (t & 7) * 8;
    async_b128(ldsOff(&Qt[x * QROW + c8]),
               &Q[(baseBS + c * WW + x) * EE + ecol + c8]);
  }
  // ---- stage V^T window: row d, columns y (key-contiguous in global) ----
  const unsigned short* vsrc = VT + (((size_t)b * HH + h) * DD) * SS;
  for (int t = tid; t < DD * 96; t += 512) {
    int d = t / 96, c8 = (t % 96) * 8;
    int j0 = c * WW + c8 - WW;
    unsigned short* dst = &VtT[d * VROW + c8];
    if (j0 >= 0 && j0 + 7 < SS) {
      async_b128(ldsOff(dst), vsrc + (size_t)d * SS + j0);
    } else {
#pragma unroll
      for (int e = 0; e < 8; ++e) {
        int j = j0 + e;
        dst[e] = (j >= 0 && j < SS) ? vsrc[(size_t)d * SS + j] : 0;
      }
    }
  }
  wait_async0();
  __syncthreads();

  const int qrow = wave * 16;
  const int half = lane >> 4;
  const int nn   = lane & 15;
  const int ar   = lane & 15;
  const int akb  = half * 16;
  const int bkb  = half * 16;

  // ---- Q A-fragments: two K=32 slices over D=64 ----
  const unsigned short* qrp = Qt + (qrow + ar) * QROW;
  v16bf qa0 = ldfrag(qrp + akb, qrp + akb + 8);
  v16bf qa1 = ldfrag(qrp + 32 + akb, qrp + 32 + akb + 8);

  float rmax[8], rsum[8];
#pragma unroll
  for (int r = 0; r < 8; ++r) { rmax[r] = -1e30f; rsum[r] = 0.0f; }
  v8f cacc[4] = {};

  unsigned short* psw = Ps + wave * 16 * PROW;

  for (int y0 = 0; y0 < KWIN; y0 += 32) {
    // ---- scores = Q (16x64) * K^T (64x32): two 16x16 f32 blocks ----
    const unsigned short* kr0 = Kt + (y0 + nn) * KROW;
    const unsigned short* kr1 = Kt + (y0 + 16 + nn) * KROW;
    v8f s0 = {}, s1 = {};
#pragma unroll
    for (int t2 = 0; t2 < 2; ++t2) {
      v16bf kf0 = ldfrag(kr0 + t2 * 32 + bkb, kr0 + t2 * 32 + bkb + 8);
      v16bf kf1 = ldfrag(kr1 + t2 * 32 + bkb, kr1 + t2 * 32 + bkb + 8);
      v16bf qa  = (t2 == 0) ? qa0 : qa1;
      s0 = __builtin_amdgcn_wmma_f32_16x16x32_bf16(false, qa, false, kf0,
                                                   (short)0, s0, false, false);
      s1 = __builtin_amdgcn_wmma_f32_16x16x32_bf16(false, qa, false, kf1,
                                                   (short)0, s1, false, false);
    }

    // ---- band + attention_mask ----
    const int y_0 = y0 + nn, y_1 = y0 + 16 + nn;
    const int j0 = c * WW + y_0 - WW, j1 = c * WW + y_1 - WW;
    const bool jv0 = (j0 >= 0) && (j0 < SS);
    const bool jv1 = (j1 >= 0) && (j1 < SS);
    const float fm0 = jv0 ? ((am[baseBS + j0] != 0) ? -10000.0f : 0.0f) : 0.0f;
    const float fm1 = jv1 ? ((am[baseBS + j1] != 0) ? -10000.0f : 0.0f) : 0.0f;

    float pscale[8];
#pragma unroll
    for (int r = 0; r < 8; ++r) {
      int x = qrow + r + 8 * half;                 // chunk-local query index
      bool v0 = jv0 && (y_0 - WW - x >= -WW) && (y_0 - WW - x <= WW);
      bool v1 = jv1 && (y_1 - WW - x >= -WW) && (y_1 - WW - x <= WW);
      s0[r] = v0 ? (s0[r] + fm0) : -1e30f;
      s1[r] = v1 ? (s1[r] + fm1) : -1e30f;

      float m = fmaxf(s0[r], s1[r]);
      m = fmaxf(m, __shfl_xor(m, 1, 32));
      m = fmaxf(m, __shfl_xor(m, 2, 32));
      m = fmaxf(m, __shfl_xor(m, 4, 32));
      m = fmaxf(m, __shfl_xor(m, 8, 32));
      float newmax = fmaxf(rmax[r], m);
      float sc = __expf(rmax[r] - newmax);
      rmax[r] = newmax;
      rsum[r] *= sc;
      pscale[r] = sc;

      float p0 = (s0[r] > -1e29f) ? __expf(s0[r] - newmax) : 0.0f;
      float p1 = (s1[r] > -1e29f) ? __expf(s1[r] - newmax) : 0.0f;
      rsum[r] += p0 + p1;

      int mrow = r + 8 * half;
      psw[mrow * PROW + nn]      = f2bf(p0);
      psw[mrow * PROW + 16 + nn] = f2bf(p1);
    }

#pragma unroll
    for (int t = 0; t < 4; ++t)
#pragma unroll
      for (int r = 0; r < 8; ++r) cacc[t][r] *= pscale[r];

    // ---- P back as A-fragment (same-wave LDS round trip) ----
    const unsigned short* prp = psw + ar * PROW;
    v16bf pf = ldfrag(prp + akb, prp + akb + 8);

    // ---- ctx += P (16x32) * V (32x64): 4 WMMAs over D blocks of 16 ----
#pragma unroll
    for (int t = 0; t < 4; ++t) {
      const unsigned short* vrp = VtT + (t * 16 + nn) * VROW + y0 + bkb;
      v16bf vf = ldfrag(vrp, vrp + 8);
      cacc[t] = __builtin_amdgcn_wmma_f32_16x16x32_bf16(false, pf, false, vf,
                                                        (short)0, cacc[t],
                                                        false, false);
    }
  }

  // ---- finalize ----
#pragma unroll
  for (int r = 0; r < 8; ++r) {
    float s = rsum[r];
    s += __shfl_xor(s, 1, 32);
    s += __shfl_xor(s, 2, 32);
    s += __shfl_xor(s, 4, 32);
    s += __shfl_xor(s, 8, 32);
    float inv = (s > 0.0f) ? (1.0f / s) : 0.0f;
    int mrow = qrow + r + 8 * half;
    size_t orow = (baseBS + c * WW + mrow) * EE + ecol;
#pragma unroll
    for (int t = 0; t < 4; ++t)
      out[orow + t * 16 + nn] = cacc[t][r] * inv;
  }
}

// ---------------------------------------------------------------------------
// Launch
// ---------------------------------------------------------------------------
extern "C" void kernel_launch(void* const* d_in, const int* in_sizes, int n_in,
                              void* d_out, int out_size, void* d_ws, size_t ws_size,
                              hipStream_t stream) {
  const float* hs = (const float*)d_in[0];
  const int*   am = (const int*)d_in[1];
  const float* qw = (const float*)d_in[2];
  const float* qb = (const float*)d_in[3];
  const float* kw = (const float*)d_in[4];
  const float* kb = (const float*)d_in[5];
  const float* vw = (const float*)d_in[6];
  const float* vb = (const float*)d_in[7];
  float* out = (float*)d_out;

  const size_t NTOK = (size_t)BB * SS;           // 8192
  unsigned short* Xbf = (unsigned short*)d_ws;
  unsigned short* Wqb = Xbf + NTOK * EE;
  unsigned short* Wkb = Wqb + (size_t)EE * EE;
  unsigned short* Wvb = Wkb + (size_t)EE * EE;
  unsigned short* Qbf = Wvb + (size_t)EE * EE;
  unsigned short* Kbf = Qbf + NTOK * EE;
  unsigned short* VTb = Kbf + NTOK * EE;         // [B,H,D,S]

  const int nH4 = (int)(NTOK * EE / 4);
  wf_cvt_bf16<<<(nH4 + 255) / 256, 256, 0, stream>>>((const float4*)hs,
                                                     (uint2*)Xbf, nH4);
  const int nW4 = EE * EE / 4;
  wf_cvt_bf16<<<nW4 / 256, 256, 0, stream>>>((const float4*)qw, (uint2*)Wqb, nW4);
  wf_cvt_bf16<<<nW4 / 256, 256, 0, stream>>>((const float4*)kw, (uint2*)Wkb, nW4);
  wf_cvt_bf16<<<nW4 / 256, 256, 0, stream>>>((const float4*)vw, (uint2*)Wvb, nW4);

  dim3 gGemm((unsigned)(NTOK / GEMM_M), EE / GEMM_N, 3);
  wf_qkv_gemm<<<gGemm, 256, 0, stream>>>(Xbf, Wqb, Wkb, Wvb, qb, kb, vb,
                                         Qbf, Kbf, VTb);

  // LDS: K (768*72) + V^T (64*776) + Q (256*72) + P (16*16*40), bf16 each
  const size_t smem = ((size_t)KWIN * KROW + (size_t)DD * VROW +
                       (size_t)WW * QROW + (size_t)16 * 16 * PROW) *
                      sizeof(unsigned short);

  dim3 gAttn(NCC, HH, BB);
  wf_banded_attn<<<gAttn, 512, smem, stream>>>(Qbf, Kbf, VTb, am, out);
}
